// LossEvaluator_41583873360401
// MI455X (gfx1250) — matrix operand order repaired and checked by reading, not compiled, not run on hardware
//
#include <hip/hip_runtime.h>
#include <hip/hip_bf16.h>
#include <math.h>

typedef __attribute__((ext_vector_type(16))) _Float16 v16h;
typedef __attribute__((ext_vector_type(8)))  _Float16 v8h;
typedef __attribute__((ext_vector_type(8)))  float    v8f;

static constexpr int Bc = 32, Sc = 128, Nc = 32, NCc = 8, MTc = 8, ZDc = 64, Fc = 64, CHc = 128;
static constexpr int MROWS = Bc * Sc * Nc;           // 131072
static constexpr int K1 = 224;                       // 193 padded to 7*32
static constexpr int K2 = 192;                       // CH + ZD
static constexpr float LOG_SQRT_2PI = 0.9189385332046727f;

// ---------------------------------------------------------------- prep ------
__global__ void k_prep(const float* __restrict__ bd, const float* __restrict__ z0,
                       const float* __restrict__ rates, const float* __restrict__ acr,
                       const float* __restrict__ tm_in,
                       const float* __restrict__ ots, const float* __restrict__ otr,
                       _Float16* __restrict__ X1, _Float16* __restrict__ Hh,
                       float* __restrict__ close_rates_ws, float* __restrict__ closeout_ws)
{
  int r = blockIdx.x * blockDim.x + threadIdx.x;
  if (r >= MROWS) return;
  int bs = r / Nc;
  int b  = bs / Sc;
  const float* rr = rates + (size_t)bs * NCc * 2;
  const float* ar = acr   + (size_t)bs * NCc;
  const float* po = ots   + (size_t)r * NCc * MTc;
  const float* pr = otr   + (size_t)r * NCc * MTc;
  float tm = tm_in[r];
  float inv_tm = 1.0f / tm;

  _Float16* x = X1 + (size_t)r * K1;
  for (int j = 0; j < Fc; ++j) x[j] = (_Float16)bd[(size_t)r * Fc + j];

  float total_used = 0.f, pl_sum = 0.f;
  for (int c = 0; c < NCc; ++c) {
    float a = ar[c];
    float first = po[c * MTc + (MTc - 1)];
    float crate = first > 0.f ? rr[c * 2 + 0] : (first < 0.f ? rr[c * 2 + 1] : 1.0f);
    close_rates_ws[(size_t)r * NCc + c] = crate;
    for (int t = 0; t < MTc; ++t) {
      float s  = po[c * MTc + t];
      float rt = pr[c * MTc + t];
      float accs = s / a;                 // acc_sizes; otm == accs (LEV = 1)
      total_used += fabsf(accs);
      float nzr = (rt != 0.f) ? rt : 1.0f;
      pl_sum += accs * (1.0f - crate / nzr);
      x[Fc + c * MTc + t] = (_Float16)(accs * inv_tm);
    }
  }
  x[Fc + 64] = (_Float16)((tm - total_used) * inv_tm);

  const float* rlast = rates + ((size_t)b * Sc + (Sc - 1)) * NCc * 2;
  for (int c = 0; c < NCc; ++c) {
    float denom = 0.5f * (rlast[c * 2 + 0] + rlast[c * 2 + 1]);
    for (int t = 0; t < MTc; ++t)
      x[Fc + 65 + c * MTc + t] = (_Float16)(pr[c * MTc + t] / denom);
  }
  for (int j = Fc + 129; j < K1; ++j) x[j] = (_Float16)0.f;

  closeout_ws[r] = ((tm + pl_sum) < 0.5f * total_used) ? 1.f : 0.f;

  _Float16* h = Hh + (size_t)r * K2 + CHc;
  for (int j = 0; j < ZDc; ++j) h[j] = (_Float16)z0[(size_t)r * ZDc + j];
}

// ----------------------------------------------------- weight packing -------
// dst[((ktile*ntiles + ntile)*32 + lane)*16 + e] =
//   W[(ktile*32 + (lane>=16)*16 + e)*Nout + ntile*16 + lane%16]
__global__ void k_pack_w(const float* __restrict__ src, _Float16* __restrict__ dst,
                         int K, int Korig, int Nout)
{
  int total = (K / 32) * (Nout / 16) * 512;
  int idx = blockIdx.x * blockDim.x + threadIdx.x;
  if (idx >= total) return;
  int e = idx & 15, lane = (idx >> 4) & 31, rest = idx >> 9;
  int ntiles = Nout >> 4;
  int ntile = rest % ntiles, ktile = rest / ntiles;
  int kk = ktile * 32 + (lane >> 4) * 16 + e;
  int nn = ntile * 16 + (lane & 15);
  float v = (kk < Korig) ? src[(size_t)kk * Nout + nn] : 0.f;
  dst[idx] = (_Float16)v;
}

__global__ void k_pack_wcat(const float* __restrict__ Wg, const float* __restrict__ Wp,
                            const float* __restrict__ Wsx, _Float16* __restrict__ dst)
{
  int total = 4 * 12 * 512;               // K=128 -> 4 ktiles, Nout=192 -> 12 ntiles
  int idx = blockIdx.x * blockDim.x + threadIdx.x;
  if (idx >= total) return;
  int e = idx & 15, lane = (idx >> 4) & 31, rest = idx >> 9;
  int ntile = rest % 12, ktile = rest / 12;
  int kk = ktile * 32 + (lane >> 4) * 16 + e;
  int nn = ntile * 16 + (lane & 15);
  float v = (nn < 64) ? Wg[(size_t)kk * 64 + nn]
          : (nn < 128) ? Wp[(size_t)kk * 64 + nn - 64]
                       : Wsx[(size_t)kk * 64 + nn - 128];
  dst[idx] = (_Float16)v;
}

__global__ void k_cat_bias(const float* __restrict__ bg, const float* __restrict__ bp,
                           const float* __restrict__ bsx, float* __restrict__ dst)
{
  int i = threadIdx.x;                    // launched with 192 threads
  dst[i] = (i < 64) ? bg[i] : (i < 128) ? bp[i - 64] : bsx[i - 128];
}

// ------------------------------------------------------------- WMMA GEMM ----
// One wave computes a 16 x (16*NT) tile: per k-step one A-fragment load feeds
// NT WMMAs (4x less A traffic than 16x16-per-wave). f32 accumulate.
template <int NT>
__global__ void k_gemm(const _Float16* __restrict__ A, int lda,
                       const _Float16* __restrict__ Wpk, const float* __restrict__ bias,
                       int K, int Nout,
                       float* __restrict__ Cf, int ldc,
                       _Float16* __restrict__ Ch, int ldch, int relu)
{
  int wave    = blockIdx.x * (blockDim.x >> 5) + (threadIdx.x >> 5);
  int ntiles  = Nout >> 4;
  int ngroups = ntiles / NT;
  int tng = wave % ngroups;               // group of NT consecutive N-tiles
  int tmi = wave / ngroups;
  if (tmi >= (MROWS >> 4)) return;        // wave-uniform; EXEC stays all-ones for WMMA
  int lane = threadIdx.x & 31;
  int half = lane >> 4;
  int mrow = tmi * 16 + (lane & 15);
  const _Float16* arow = A + (size_t)mrow * lda;

  v8f c[NT];
#pragma unroll
  for (int j = 0; j < NT; ++j) c[j] = (v8f){};

  int kt = 0;
  for (int k0 = 0; k0 < K; k0 += 32, ++kt) {
    v8h lo = *(const v8h*)(arow + k0 + half * 16);       // K = k0 + half*16 .. +7
    v8h hi = *(const v8h*)(arow + k0 + half * 16 + 8);   // K = k0+half*16+8 .. +7
    v16h a = __builtin_shufflevector(lo, hi, 0,1,2,3,4,5,6,7,8,9,10,11,12,13,14,15);
    const _Float16* wrow = Wpk + (((size_t)kt * ntiles + tng * NT) * 32 + lane) * 16;
#pragma unroll
    for (int j = 0; j < NT; ++j) {
      v16h bfrag = *(const v16h*)(wrow + (size_t)j * 512);
      c[j] = __builtin_amdgcn_wmma_f32_16x16x32_f16(false, a, false, bfrag,
                                                    (short)0, c[j], false, false);
    }
  }

#pragma unroll
  for (int j = 0; j < NT; ++j) {
    int ng = (tng * NT + j) * 16 + (lane & 15);
    float bv = bias[ng];
#pragma unroll
    for (int rr = 0; rr < 8; ++rr) {
      int mg = tmi * 16 + half * 8 + rr;
      float v = c[j][rr] + bv;
      if (relu) v = fmaxf(v, 0.f);
      if (Cf) Cf[(size_t)mg * ldc + ng] = v;
      else    Ch[(size_t)mg * ldch + ng] = (_Float16)v;
    }
  }
}

// ------------------------------------------------------------- latent -------
__global__ void k_latent(const float* __restrict__ G, const float* __restrict__ ZLIN,
                         const float* __restrict__ eps, float* __restrict__ z_out,
                         _Float16* __restrict__ ZH, float* __restrict__ zlogp)
{
  int r = blockIdx.x * blockDim.x + threadIdx.x;
  if (r >= MROWS) return;
  const float* g  = G    + (size_t)r * 192;   // [gate_lin | p_lin | s_lin]
  const float* zl = ZLIN + (size_t)r * 64;
  const float* ep = eps  + (size_t)r * 64;
  float acc = 0.f;
  for (int d = 0; d < ZDc; ++d) {
    float gate = 1.f / (1.f + expf(-g[d]));
    float zloc = (1.f - gate) * zl[d] + gate * g[64 + d];
    float sp = g[128 + d];
    float zscale = fmaxf(sp, 0.f) + log1pf(expf(-fabsf(sp))) + 1e-4f;  // softplus + eps
    float zv = zloc + zscale * ep[d];
    z_out[(size_t)r * 64 + d] = zv;
    ZH[(size_t)r * 64 + d]   = (_Float16)zv;
    float t = (zv - zloc) / zscale;
    acc += -0.5f * t * t - logf(zscale) - LOG_SQRT_2PI;
  }
  zlogp[r] = acc;
}

// ------------------------------------------------------------- trading ------
__global__ void k_trade(const float* __restrict__ EM, const float* __restrict__ uex,
                        const float* __restrict__ rates, const float* __restrict__ acr,
                        const float* __restrict__ tm_in,
                        const float* __restrict__ ots_in, const float* __restrict__ otr_in,
                        const float* __restrict__ crws, const float* __restrict__ cows,
                        const float* __restrict__ zlogp,
                        float* __restrict__ tm_out, float* __restrict__ ots_out,
                        float* __restrict__ otr_out, float* __restrict__ logp_out)
{
  int r = blockIdx.x * blockDim.x + threadIdx.x;
  if (r >= MROWS) return;
  int bs = r / Nc;
  const float* rr = rates + (size_t)bs * NCc * 2;
  const float* ar = acr   + (size_t)bs * NCc;
  const float* em = EM    + (size_t)r * 16;
  const float* u  = uex   + (size_t)r * NCc;
  const float* po = ots_in + (size_t)r * 64;
  const float* pr = otr_in + (size_t)r * 64;
  bool closeout = cows[r] > 0.5f;
  float tm = tm_in[r];

  float frac[NCc], orate[NCc], elp[NCc], pos[NCc], umarg[NCc];
  bool  emask[NCc], opp[NCc], ign[NCc], addtr[NCc];
  float used_sum = 0.f;
  for (int c = 0; c < NCc; ++c) {
    float lg = em[2 * c];
    frac[c] = tanhf(em[2 * c + 1]);
    bool es = u[c] < 1.f / (1.f + expf(-lg));
    emask[c] = es;
    float la = log1pf(expf(-fabsf(lg)));
    elp[c] = es ? (fminf(lg, 0.f) - la) : (fminf(-lg, 0.f) - la);
    float first = po[c * MTc + MTc - 1];
    float last  = po[c * MTc];
    opp[c]   = frac[c] * first <= 0.f;
    ign[c]   = (!opp[c]) && (last != 0.f);
    addtr[c] = (!opp[c]) && (last == 0.f);
    orate[c] = frac[c] > 0.f ? rr[2 * c + 1] : (frac[c] < 0.f ? rr[2 * c] : 0.f);
    float ps = 0.f;
    for (int t = 0; t < MTc; ++t) ps += po[c * MTc + t] / ar[c];
    pos[c] = ps;
    umarg[c] = fabsf(ps) / ar[c];
    used_sum += umarg[c];
  }

  for (int i = 0; i < NCc; ++i) {
    float avail = (tm - used_sum) + (opp[i] ? umarg[i] : 0.f);
    float ex = frac[i] * avail * ar[i];
    if (!emask[i]) ex = 0.f;
    if (closeout)  ex = -pos[i];

    float cs[MTc], cr[MTc];
    for (int t = 0; t < MTc; ++t) { cs[t] = po[i * MTc + t]; cr[t] = pr[i * MTc + t]; }
    if (addtr[i]) {
      for (int t = 0; t < MTc - 1; ++t) { cs[t] = cs[t + 1]; cr[t] = cr[t + 1]; }
      cs[MTc - 1] = ex; cr[MTc - 1] = orate[i];
    }
    float left_ex = opp[i] ? ex : 0.f;
    float lc[MTc], rc[MTc], run = left_ex;
    for (int t = 0; t < MTc; ++t) { lc[t] = run; run += cs[t]; rc[t] = run; }

    float crate = crws[(size_t)r * NCc + i];
    float plsum = 0.f;
    for (int t = 0; t < MTc; ++t) {
      bool red = lc[t] * rc[t] < 0.f;
      bool clm = rc[t] * left_ex >= 0.f;
      float closed = clm ? cs[t] : 0.f;
      float ncs = clm ? 0.f : cs[t];
      float ncr = clm ? 0.f : cr[t];
      if (red) { closed = lc[t]; ncs = rc[t]; }
      float origr = pr[i * MTc + t];
      float nzr = (origr != 0.f) ? origr : 1.f;
      plsum += fabsf(closed) / ar[i] * (1.f - crate / nzr);
      if (t == MTc - 1 && clm) { ncs = rc[t]; ncr = orate[i]; }   // add_opp
      cs[t] = ncs; cr[t] = ncr;
    }
    tm += plsum;
    for (int t = 0; t < MTc; ++t) {
      ots_out[(size_t)r * 64 + i * MTc + t] = cs[t];
      otr_out[(size_t)r * 64 + i * MTc + t] = cr[t];
    }
  }
  tm_out[r] = tm;
  float lp = zlogp[r];
  for (int c = 0; c < NCc; ++c) if (!(closeout || ign[c])) lp += elp[c];
  logp_out[r] = lp;
}

// --------------------------------------------------------------- losses -----
__global__ void k_loss(const float* __restrict__ tm_out, const float* __restrict__ logp,
                       float* __restrict__ losses)
{
  int bs = blockIdx.x * blockDim.x + threadIdx.x;
  if (bs >= Bc * Sc) return;
  float costs[Nc]; float mean = 0.f;
  for (int n = 0; n < Nc; ++n) {
    float cst = -logf(tm_out[(size_t)bs * Nc + n]);
    costs[n] = cst; mean += cst;
  }
  mean *= (1.f / Nc);
  for (int n = 0; n < Nc; ++n)
    losses[(size_t)bs * Nc + n] = logp[(size_t)bs * Nc + n] * (costs[n] - mean) + costs[n];
}

// -------------------------------------------------------------- launch ------
extern "C" void kernel_launch(void* const* d_in, const int* in_sizes, int n_in,
                              void* d_out, int out_size, void* d_ws, size_t ws_size,
                              hipStream_t stream)
{
  (void)in_sizes; (void)n_in; (void)out_size; (void)ws_size;
  const float* bd    = (const float*)d_in[0];
  const float* z0    = (const float*)d_in[1];
  const float* rates = (const float*)d_in[2];
  const float* acr   = (const float*)d_in[3];
  const float* tm    = (const float*)d_in[4];
  const float* ots   = (const float*)d_in[5];
  const float* otr   = (const float*)d_in[6];
  const float* eps   = (const float*)d_in[7];
  const float* uex   = (const float*)d_in[8];
  const float* Wcnn  = (const float*)d_in[9];
  const float* bcnn  = (const float*)d_in[10];
  const float* Wh    = (const float*)d_in[11];
  const float* bh    = (const float*)d_in[12];
  const float* Wg    = (const float*)d_in[13];
  const float* bg    = (const float*)d_in[14];
  const float* Wp    = (const float*)d_in[15];
  const float* bp    = (const float*)d_in[16];
  const float* Wlin  = (const float*)d_in[17];
  const float* blin  = (const float*)d_in[18];
  const float* Wsw   = (const float*)d_in[19];
  const float* bsw   = (const float*)d_in[20];
  const float* We    = (const float*)d_in[21];
  const float* be    = (const float*)d_in[22];

  float* out     = (float*)d_out;
  float* losses  = out;
  float* z_out   = out + MROWS;
  float* tm_out  = z_out + (size_t)MROWS * 64;
  float* ots_out = tm_out + MROWS;
  float* otr_out = ots_out + (size_t)MROWS * 64;

  char* ws = (char*)d_ws;
  const size_t SZ_X1 = (size_t)MROWS * K1 * 2;     // 58,720,256
  const size_t SZ_H  = (size_t)MROWS * K2 * 2;     // 50,331,648
  const size_t OFF_X1 = 0;
  const size_t OFF_H  = SZ_X1;
  const size_t OFF_G  = 0;                          // G (M x 192 f32) aliases X1+H after they are dead
  const size_t OFF_HT   = SZ_X1 + SZ_H;             // M x 128 f16
  const size_t OFF_ZLIN = OFF_HT   + (size_t)MROWS * 128 * 2;
  const size_t OFF_ZH   = OFF_ZLIN + (size_t)MROWS * 64 * 4;
  const size_t OFF_EM   = OFF_ZH   + (size_t)MROWS * 64 * 2;
  const size_t OFF_ZLP  = OFF_EM   + (size_t)MROWS * 16 * 4;
  const size_t OFF_LP   = OFF_ZLP  + (size_t)MROWS * 4;
  const size_t OFF_CR   = OFF_LP   + (size_t)MROWS * 4;
  const size_t OFF_CO   = OFF_CR   + (size_t)MROWS * 8 * 4;
  const size_t OFF_WCNN = OFF_CO   + (size_t)MROWS * 4;
  const size_t OFF_WH   = OFF_WCNN + 7 * 8 * 512 * 2;
  const size_t OFF_WCAT = OFF_WH   + 6 * 8 * 512 * 2;
  const size_t OFF_WLIN = OFF_WCAT + 4 * 12 * 512 * 2;
  const size_t OFF_WE   = OFF_WLIN + 2 * 4 * 512 * 2;
  const size_t OFF_BCAT = OFF_WE   + 2 * 1 * 512 * 2;

  _Float16* X1h   = (_Float16*)(ws + OFF_X1);
  _Float16* Hh    = (_Float16*)(ws + OFF_H);
  float*    Gf    = (float*)   (ws + OFF_G);
  _Float16* HTh   = (_Float16*)(ws + OFF_HT);
  float*    ZLINf = (float*)   (ws + OFF_ZLIN);
  _Float16* ZHh   = (_Float16*)(ws + OFF_ZH);
  float*    EMf   = (float*)   (ws + OFF_EM);
  float*    zlogp = (float*)   (ws + OFF_ZLP);
  float*    logp  = (float*)   (ws + OFF_LP);
  float*    crws  = (float*)   (ws + OFF_CR);
  float*    cows  = (float*)   (ws + OFF_CO);
  _Float16* Wcnn16= (_Float16*)(ws + OFF_WCNN);
  _Float16* Wh16  = (_Float16*)(ws + OFF_WH);
  _Float16* Wcat16= (_Float16*)(ws + OFF_WCAT);
  _Float16* Wlin16= (_Float16*)(ws + OFF_WLIN);
  _Float16* We16  = (_Float16*)(ws + OFF_WE);
  float*    bcat  = (float*)   (ws + OFF_BCAT);

  // weight packing (tiny, L2-resident afterwards)
  k_pack_w<<<(7 * 8 * 512 + 255) / 256, 256, 0, stream>>>(Wcnn, Wcnn16, 224, 193, 128);
  k_pack_w<<<(6 * 8 * 512 + 255) / 256, 256, 0, stream>>>(Wh, Wh16, 192, 192, 128);
  k_pack_wcat<<<(4 * 12 * 512 + 255) / 256, 256, 0, stream>>>(Wg, Wp, Wsw, Wcat16);
  k_pack_w<<<(2 * 4 * 512 + 255) / 256, 256, 0, stream>>>(Wlin, Wlin16, 64, 64, 64);
  k_pack_w<<<(2 * 1 * 512 + 255) / 256, 256, 0, stream>>>(We, We16, 64, 64, 16);
  k_cat_bias<<<1, 192, 0, stream>>>(bg, bp, bsw, bcat);

  k_prep<<<MROWS / 256, 256, 0, stream>>>(bd, z0, rates, acr, tm, ots, otr,
                                          X1h, Hh, crws, cows);

  // GEMM1: X1 (224) -> H[:, :128], relu, f16 out.  8192 M-tiles x 2 N-groups
  k_gemm<4><<<16384 / 8, 256, 0, stream>>>(X1h, K1, Wcnn16, bcnn, 224, 128,
                                           (float*)nullptr, 0, Hh, K2, 1);
  // GEMM2: H (192) -> HT (128), relu, f16 out
  k_gemm<4><<<16384 / 8, 256, 0, stream>>>(Hh, K2, Wh16, bh, 192, 128,
                                           (float*)nullptr, 0, HTh, 128, 1);
  // GEMM4 (before GEMM3 — G aliases H): z0 (H cols 128..191) @ Wlin -> ZLIN f32
  k_gemm<4><<<8192 / 8, 256, 0, stream>>>(Hh + CHc, K2, Wlin16, blin, 64, 64,
                                          ZLINf, 64, (_Float16*)nullptr, 0, 0);
  // GEMM3: HT (128) @ [Wg|Wp|Ws] -> G (M x 192) f32.  8192 x 3 groups
  k_gemm<4><<<24576 / 8, 256, 0, stream>>>(HTh, 128, Wcat16, bcat, 128, 192,
                                           Gf, 192, (_Float16*)nullptr, 0, 0);

  k_latent<<<MROWS / 256, 256, 0, stream>>>(Gf, ZLINf, eps, z_out, ZHh, zlogp);

  // GEMM5: z (64) @ We -> EM (M x 16) f32
  k_gemm<1><<<8192 / 8, 256, 0, stream>>>(ZHh, 64, We16, be, 64, 16,
                                          EMf, 16, (_Float16*)nullptr, 0, 0);

  k_trade<<<MROWS / 256, 256, 0, stream>>>(EMf, uex, rates, acr, tm, ots, otr,
                                           crws, cows, zlogp,
                                           tm_out, ots_out, otr_out, logp);

  k_loss<<<(Bc * Sc + 255) / 256, 256, 0, stream>>>(tm_out, logp, losses);
}
